// LSTM_26740466385735
// MI455X (gfx1250) — matrix operand order repaired and checked by reading, not compiled, not run on hardware
//
#include <hip/hip_runtime.h>

// ============================================================================
// LSTM (T=512, B=64, I=H=1024) for gfx1250 / MI455X, wave32 + WMMA bf16.
//
//   Phase A (parallel):  G[T*B,4H] = bf16(x) @ bf16(Wcat_i)^T            (WMMA)
//   Phase B (sequential): per step t, pre = G[t] + h_{t-1} @ Wcat_h^T + b (WMMA)
//                         gate activations + cell update, h double-buffered.
//   Phase B stages h chunks with GLOBAL_LOAD_ASYNC_TO_LDS_B128 (ASYNCcnt),
//   double-buffered in LDS so chunk k+1 streams while chunk k is in WMMA.
//
// Workspace layout (bytes), total ~529 MB:
//   [0,8M)        Wi_bf   bf16 [4096][1024]   (gates i,f,g,o input weights)
//   [8M,16M)      Wh_bf   bf16 [4096][1024]   (hidden weights)
//   +16K          btot    f32  [4096]         (b_i* + b_h*)
//   +128K +128K   h ping/pong  bf16 [64][1024]
//   +256K         c state f32  [64][1024]
//   +512M         G       f32  [32768][4096]
// ============================================================================

typedef __attribute__((ext_vector_type(16))) __bf16 bf16x16;
typedef __attribute__((ext_vector_type(8)))  float  f32x8;

union FragB { uint4 u[2]; bf16x16 v; };

__device__ __forceinline__ unsigned int pack_bf2(float a, float b) {
  union { float f; unsigned int u; } x, y;
  x.f = a; y.f = b;
  unsigned int ra = (x.u + 0x7FFFu + ((x.u >> 16) & 1u)) >> 16;
  unsigned int rb = (y.u + 0x7FFFu + ((y.u >> 16) & 1u)) >> 16;
  return (ra & 0xFFFFu) | (rb << 16);
}

__device__ __forceinline__ unsigned short f2bf1(float a) {
  union { float f; unsigned int u; } x; x.f = a;
  return (unsigned short)((x.u + 0x7FFFu + ((x.u >> 16) & 1u)) >> 16);
}

__device__ __forceinline__ float sigmoid_f(float x) {
  return 1.0f / (1.0f + __expf(-x));
}
__device__ __forceinline__ float tanh_f(float x) {
  float e = __expf(2.0f * x);          // inf for large x -> 1; 0 for small -> -1
  return 1.0f - 2.0f / (e + 1.0f);
}

// Async DMA: 16 bytes global -> LDS, tracked by ASYNCcnt (no VGPR round trip).
// lds_off is the byte offset within the workgroup's LDS allocation (low 32
// bits of the flat address of a __shared__ object per the aperture rules).
__device__ __forceinline__ void async_b128_to_lds(unsigned int lds_off,
                                                  const void* gaddr) {
  asm volatile("global_load_async_to_lds_b128 %0, %1, off"
               :: "v"(lds_off), "v"(gaddr) : "memory");
}

// ---------------------------------------------------------------------------
// Pack 4 fp32 [1024][1024] weight matrices into one bf16 [4096][1024] block.
// dst viewed as uint (2 bf16 per uint): 4*1024*512 uints.
// ---------------------------------------------------------------------------
__global__ __launch_bounds__(256)
void pack_w4(const float* __restrict__ w0, const float* __restrict__ w1,
             const float* __restrict__ w2, const float* __restrict__ w3,
             unsigned int* __restrict__ dst) {
  size_t idx = (size_t)blockIdx.x * 256 + threadIdx.x;   // < 4*1024*512
  size_t g = idx / (1024u * 512u);
  size_t r = idx % (1024u * 512u);
  const float* s = (g == 0) ? w0 : (g == 1) ? w1 : (g == 2) ? w2 : w3;
  s += r * 2;
  dst[idx] = pack_bf2(s[0], s[1]);
}

__global__ __launch_bounds__(256)
void pack_bias(const float* bi0, const float* bi1, const float* bi2, const float* bi3,
               const float* bh0, const float* bh1, const float* bh2, const float* bh3,
               float* __restrict__ btot) {
  int idx = blockIdx.x * 256 + threadIdx.x;   // < 4096
  int g = idx >> 10, j = idx & 1023;
  const float* bi = (g == 0) ? bi0 : (g == 1) ? bi1 : (g == 2) ? bi2 : bi3;
  const float* bh = (g == 0) ? bh0 : (g == 1) ? bh1 : (g == 2) ? bh2 : bh3;
  btot[idx] = bi[j] + bh[j];
}

__global__ __launch_bounds__(256)
void init_state(const float* __restrict__ h0, const float* __restrict__ c0,
                unsigned int* __restrict__ hbf, float* __restrict__ c_ws) {
  int idx = blockIdx.x * 256 + threadIdx.x;   // < 65536
  c_ws[idx] = c0[idx];
  if (idx < 32768) hbf[idx] = pack_bf2(h0[2 * idx], h0[2 * idx + 1]);
}

// ---------------------------------------------------------------------------
// Phase A: G = bf16(x) @ Wi_bf^T.  M=32768, N=4096, K=1024.
// Grid (32, 256): block tile 128(M) x 128(N), K-chunks of 32 staged in LDS.
// 8 waves: wave = (wn<4) x (wm<2); each wave: 4 m-tiles x 2 n-tiles (64x32).
// ---------------------------------------------------------------------------
__global__ __launch_bounds__(256)
void lstm_input_gemm(const float* __restrict__ x,            // [32768][1024]
                     const unsigned short* __restrict__ Wi,  // bf16 [4096][1024]
                     float* __restrict__ G) {                // [32768][4096]
  __shared__ unsigned short As[128 * 32];   // 8 KB
  __shared__ unsigned short Bs[128 * 32];   // 8 KB

  const int tid  = threadIdx.x;
  const int bxN  = blockIdx.x;     // 0..31
  const int byM  = blockIdx.y;     // 0..255
  const int wave = tid >> 5, lane = tid & 31;
  const int lrow = lane & 15, lhalf = lane >> 4;
  const int wm = wave & 1, wn = wave >> 1;

  f32x8 acc[4][2];
#pragma unroll
  for (int mt = 0; mt < 4; ++mt)
#pragma unroll
    for (int nt = 0; nt < 2; ++nt)
#pragma unroll
      for (int i = 0; i < 8; ++i) acc[mt][nt][i] = 0.0f;

  const int r  = tid >> 1;   // 0..127 staged row
  const int hh = tid & 1;    // 16-element half

  for (int kc = 0; kc < 32; ++kc) {
    // ---- stage A (x fp32 -> bf16) and B (Wi bf16) chunks into LDS ----
    {
      const float* xp = x + (size_t)(byM * 128 + r) * 1024 + kc * 32 + hh * 16;
      float4 f0 = ((const float4*)xp)[0];
      float4 f1 = ((const float4*)xp)[1];
      float4 f2 = ((const float4*)xp)[2];
      float4 f3 = ((const float4*)xp)[3];
      uint4 p0, p1;
      p0.x = pack_bf2(f0.x, f0.y); p0.y = pack_bf2(f0.z, f0.w);
      p0.z = pack_bf2(f1.x, f1.y); p0.w = pack_bf2(f1.z, f1.w);
      p1.x = pack_bf2(f2.x, f2.y); p1.y = pack_bf2(f2.z, f2.w);
      p1.z = pack_bf2(f3.x, f3.y); p1.w = pack_bf2(f3.z, f3.w);
      uint4* ad = (uint4*)&As[r * 32 + hh * 16];
      ad[0] = p0; ad[1] = p1;

      const unsigned short* wp = Wi + (size_t)(bxN * 128 + r) * 1024 + kc * 32 + hh * 16;
      uint4 q0 = ((const uint4*)wp)[0];
      uint4 q1 = ((const uint4*)wp)[1];
      uint4* bd = (uint4*)&Bs[r * 32 + hh * 16];
      bd[0] = q0; bd[1] = q1;
    }
    __syncthreads();

    // ---- WMMA compute from LDS ----
    FragB bfr[2];
#pragma unroll
    for (int nt = 0; nt < 2; ++nt) {
      const unsigned short* bp = &Bs[(wn * 32 + nt * 16 + lrow) * 32 + lhalf * 16];
      bfr[nt].u[0] = ((const uint4*)bp)[0];
      bfr[nt].u[1] = ((const uint4*)bp)[1];
    }
#pragma unroll
    for (int mt = 0; mt < 4; ++mt) {
      FragB afr;
      const unsigned short* ap = &As[(wm * 64 + mt * 16 + lrow) * 32 + lhalf * 16];
      afr.u[0] = *(const uint4*)ap;
      afr.u[1] = *(const uint4*)(ap + 8);
#pragma unroll
      for (int nt = 0; nt < 2; ++nt)
        acc[mt][nt] = __builtin_amdgcn_wmma_f32_16x16x32_bf16(
            false, afr.v, false, bfr[nt].v, (short)0, acc[mt][nt], false, false);
    }
    __syncthreads();
  }

  // ---- write back (C layout: vgpr vr -> m = lhalf*8+vr, n = lrow) ----
#pragma unroll
  for (int mt = 0; mt < 4; ++mt)
#pragma unroll
    for (int nt = 0; nt < 2; ++nt) {
      int row0 = byM * 128 + wm * 64 + mt * 16 + lhalf * 8;
      int col  = bxN * 128 + wn * 32 + nt * 16 + lrow;
      float* gp = G + (size_t)row0 * 4096 + col;
#pragma unroll
      for (int vr = 0; vr < 8; ++vr) gp[(size_t)vr * 4096] = acc[mt][nt][vr];
    }
}

// ---------------------------------------------------------------------------
// Phase B: one timestep.  32 blocks; block wg owns columns [wg*32, wg*32+32)
// of H, across all 4 gates.  8 waves: wave%4 = m-tile (16 rows of B=64),
// wave/4 = gate pair {i,f} or {g,o}; 2 n-tiles each -> 4 WMMA acc per wave.
// h chunks are streamed memory->LDS with async DMA, double-buffered, so the
// next chunk is in flight while the current one feeds the WMMAs.
// ---------------------------------------------------------------------------
__global__ __launch_bounds__(256)
void lstm_step(const unsigned short* __restrict__ hprev,  // bf16 [64][1024]
               unsigned short* __restrict__ hnext,        // bf16 [64][1024]
               const unsigned short* __restrict__ Wh,     // bf16 [4096][1024]
               const float* __restrict__ G,               // [32768][4096]
               const float* __restrict__ btot,            // [4096]
               float* __restrict__ c_ws,                  // [64][1024]
               float* __restrict__ out,                   // [512][64][1024]
               int t) {
  __shared__ unsigned short Hs[2][64 * 32];   // 2 x 4 KB h chunk (ping/pong)
  __shared__ float Gb[4 * 64 * 32];           // 32 KB gate pre-activations

  const int tid  = threadIdx.x;
  const int wg   = blockIdx.x;             // 0..31
  const int wave = tid >> 5, lane = tid & 31;
  const int lrow = lane & 15, lhalf = lane >> 4;
  const int mt = wave & 3;                 // rows [mt*16, mt*16+16)
  const int gp = wave >> 2;                // gate pair

  f32x8 acc[2][2];
#pragma unroll
  for (int gi = 0; gi < 2; ++gi)
#pragma unroll
    for (int nt = 0; nt < 2; ++nt)
#pragma unroll
      for (int i = 0; i < 8; ++i) acc[gi][nt][i] = 0.0f;

  const int srow = tid >> 2;   // 0..63
  const int sq   = tid & 3;    // 8-element quarter

  // per-thread LDS byte offset of its 16-byte slot in buffer 0
  const unsigned int lds0 =
      (unsigned int)(size_t)(void*)&Hs[0][srow * 32 + sq * 8];
  const unsigned short* gsrc = hprev + (size_t)srow * 1024 + sq * 8;

  // prologue: start streaming chunk 0
  async_b128_to_lds(lds0, gsrc);

  for (int kc = 0; kc < 32; ++kc) {
    if (kc < 31) {
      // start streaming chunk kc+1 into the other buffer; its previous
      // readers finished at the trailing barrier of iteration kc-1.
      async_b128_to_lds(lds0 + ((kc + 1) & 1) * 4096, gsrc + (kc + 1) * 32);
      asm volatile("s_wait_asynccnt 0x1" ::: "memory");  // chunk kc landed
    } else {
      asm volatile("s_wait_asynccnt 0x0" ::: "memory");
    }
    __syncthreads();

    FragB afr;
    const unsigned short* ap = &Hs[kc & 1][(mt * 16 + lrow) * 32 + lhalf * 16];
    afr.u[0] = *(const uint4*)ap;
    afr.u[1] = *(const uint4*)(ap + 8);

#pragma unroll
    for (int gi = 0; gi < 2; ++gi) {
      int gate = gp * 2 + gi;
#pragma unroll
      for (int nt = 0; nt < 2; ++nt) {
        int n = gate * 1024 + wg * 32 + nt * 16 + lrow;
        const unsigned short* wp = Wh + (size_t)n * 1024 + kc * 32 + lhalf * 16;
        FragB bfr;
        bfr.u[0] = ((const uint4*)wp)[0];
        bfr.u[1] = ((const uint4*)wp)[1];
        acc[gi][nt] = __builtin_amdgcn_wmma_f32_16x16x32_bf16(
            false, afr.v, false, bfr.v, (short)0, acc[gi][nt], false, false);
      }
    }
    __syncthreads();   // all waves done with Hs[kc&1] before it is re-filled
  }

  // spill WMMA accumulators to LDS for cross-wave gate fusion
#pragma unroll
  for (int gi = 0; gi < 2; ++gi) {
    int gate = gp * 2 + gi;
#pragma unroll
    for (int nt = 0; nt < 2; ++nt) {
      int col = nt * 16 + lrow;
#pragma unroll
      for (int vr = 0; vr < 8; ++vr) {
        int b = mt * 16 + lhalf * 8 + vr;
        Gb[(gate * 64 + b) * 32 + col] = acc[gi][nt][vr];
      }
    }
  }
  __syncthreads();

  // fused activations + cell/hidden update: 64x32 elements, 8 per thread
#pragma unroll
  for (int i = 0; i < 8; ++i) {
    int idx = tid + 256 * i;        // 0..2047
    int b = idx >> 5, col = idx & 31;
    int j = wg * 32 + col;
    const float* grow = G + (size_t)(t * 64 + b) * 4096;
    float pi = Gb[(0 * 64 + b) * 32 + col] + grow[0 * 1024 + j] + btot[0 * 1024 + j];
    float pf = Gb[(1 * 64 + b) * 32 + col] + grow[1 * 1024 + j] + btot[1 * 1024 + j];
    float pg = Gb[(2 * 64 + b) * 32 + col] + grow[2 * 1024 + j] + btot[2 * 1024 + j];
    float po = Gb[(3 * 64 + b) * 32 + col] + grow[3 * 1024 + j] + btot[3 * 1024 + j];
    float it = sigmoid_f(pi);
    float ft = sigmoid_f(pf);
    float gt = tanh_f(pg);
    float ot = sigmoid_f(po);
    size_t sidx = (size_t)b * 1024 + j;
    float c = ft * c_ws[sidx] + it * gt;
    c_ws[sidx] = c;
    float h = ot * tanh_f(c);
    hnext[sidx] = f2bf1(h);
    out[(size_t)(t * 64 + b) * 1024 + j] = h;
  }
}

// ---------------------------------------------------------------------------
extern "C" void kernel_launch(void* const* d_in, const int* in_sizes, int n_in,
                              void* d_out, int out_size, void* d_ws, size_t ws_size,
                              hipStream_t stream) {
  (void)in_sizes; (void)n_in; (void)out_size; (void)ws_size;

  const float* x    = (const float*)d_in[0];
  const float* h0   = (const float*)d_in[1];
  const float* c0   = (const float*)d_in[2];
  const float* W_ii = (const float*)d_in[3];
  const float* b_ii = (const float*)d_in[4];
  const float* W_hi = (const float*)d_in[5];
  const float* b_hi = (const float*)d_in[6];
  const float* W_if = (const float*)d_in[7];
  const float* b_if = (const float*)d_in[8];
  const float* W_hf = (const float*)d_in[9];
  const float* b_hf = (const float*)d_in[10];
  const float* W_ig = (const float*)d_in[11];
  const float* b_ig = (const float*)d_in[12];
  const float* W_hg = (const float*)d_in[13];
  const float* b_hg = (const float*)d_in[14];
  const float* W_io = (const float*)d_in[15];
  const float* b_io = (const float*)d_in[16];
  const float* W_ho = (const float*)d_in[17];
  const float* b_ho = (const float*)d_in[18];
  float* out = (float*)d_out;

  char* ws = (char*)d_ws;
  const size_t OFF_WI = 0;                         // 8 MB  bf16 [4096][1024]
  const size_t OFF_WH = OFF_WI + (size_t)8  * 1024 * 1024;
  const size_t OFF_BT = OFF_WH + (size_t)8  * 1024 * 1024;   // 16 KB
  const size_t OFF_H0 = OFF_BT + 16384;            // 128 KB bf16 ping
  const size_t OFF_H1 = OFF_H0 + 131072;           // 128 KB bf16 pong
  const size_t OFF_C  = OFF_H1 + 131072;           // 256 KB fp32 cell
  const size_t OFF_G  = OFF_C  + 262144;           // 512 MB fp32 gates

  unsigned short* Wi_bf = (unsigned short*)(ws + OFF_WI);
  unsigned short* Wh_bf = (unsigned short*)(ws + OFF_WH);
  float*          btot  = (float*)(ws + OFF_BT);
  unsigned short* hbuf0 = (unsigned short*)(ws + OFF_H0);
  unsigned short* hbuf1 = (unsigned short*)(ws + OFF_H1);
  float*          c_ws  = (float*)(ws + OFF_C);
  float*          G     = (float*)(ws + OFF_G);

  // pack weights/biases to bf16 (gate order i,f,g,o)
  pack_w4<<<8192, 256, 0, stream>>>(W_ii, W_if, W_ig, W_io, (unsigned int*)Wi_bf);
  pack_w4<<<8192, 256, 0, stream>>>(W_hi, W_hf, W_hg, W_ho, (unsigned int*)Wh_bf);
  pack_bias<<<16, 256, 0, stream>>>(b_ii, b_if, b_ig, b_io,
                                    b_hi, b_hf, b_hg, b_ho, btot);
  init_state<<<256, 256, 0, stream>>>(h0, c0, (unsigned int*)hbuf0, c_ws);

  // Phase A: all input-side projections in one WMMA GEMM
  lstm_input_gemm<<<dim3(32, 256), 256, 0, stream>>>(x, Wi_bf, G);

  // Phase B: 512 sequential steps, h double-buffered
  for (int t = 0; t < 512; ++t) {
    const unsigned short* hp = (t & 1) ? hbuf1 : hbuf0;
    unsigned short*       hn = (t & 1) ? hbuf0 : hbuf1;
    lstm_step<<<32, 256, 0, stream>>>(hp, hn, Wh_bf, G, btot, c_ws, out, t);
  }
}
